// VQ_VAE_55052890800848
// MI455X (gfx1250) — matrix operand order repaired and checked by reading, not compiled, not run on hardware
//
#include <hip/hip_runtime.h>
#include <math.h>

// ---------------------------------------------------------------------------
// CDNA5 (gfx1250) VQ-VAE forward.
//  - All convs (incl. transposed convs via output-parity decomposition) are
//    implicit GEMMs on v_wmma_f32_16x16x32_bf16 (f32 accumulate).
//    128x64 block tile, 4 waves (wave32), 64x32 per wave = 8 WMMA / K-step,
//    register-software-pipelined global->LDS staging, b128 LDS fragment loads.
//  - VQ nearest-code search is a GEMM on v_wmma_f32_16x16x4_f32 (exact f32).
// ---------------------------------------------------------------------------

typedef __bf16 bf16;
typedef __attribute__((ext_vector_type(16))) __bf16 v16bf;
typedef __attribute__((ext_vector_type(8)))  __bf16 v8bf;
typedef __attribute__((ext_vector_type(8)))  float  v8f;
typedef __attribute__((ext_vector_type(2)))  float  v2f;
typedef unsigned int       u32;
typedef unsigned long long u64;

// mode 0: regular conv, weights OIHW
// mode 1: transposed-conv parity class (pa,pb): weights (Cin,Cout,4,4),
//         stride 2, pad 1, k 4; oy/ox index the quarter grid,
//         ho = 2*oy + (1-pa), wo = 2*ox + (1-pb)
__global__ __launch_bounds__(128) void vqvae_conv_wmma(
    const float* __restrict__ X, const float* __restrict__ W,
    const float* __restrict__ Bias, const float* __restrict__ Res,
    float* __restrict__ Out,
    int Cin, int Hin, int Win, int Cout, int Hout, int Wout,
    int KW, int stride, int pad, int K, int steps,
    int khkw, int kmulA, int kmulB,
    int pshift, int pmask, int wshift, int wmask,
    int mode, int pa, int pb, int in_relu, int out_relu)
{
    __shared__ alignas(16) bf16 As[128][40];   // [row][k]   80B row stride
    __shared__ alignas(16) bf16 Bst[64][40];   // [col][k]   transposed B tile

    const int tid   = threadIdx.x;
    const int lane  = tid & 31;
    const int wave  = tid >> 5;
    const int wm    = (wave >> 1) << 6;        // 0 or 64
    const int wn    = (wave & 1) << 5;         // 0 or 32
    const int lr    = lane & 15;
    const int hhalf = lane >> 4;
    const int mbase = blockIdx.x << 7;
    const int nbase = blockIdx.y << 6;
    const int HW    = Hin * Win;

    // Per-thread A row (row = mbase + tid) decomposition, K-loop invariant.
    const int gmA  = mbase + tid;
    const int nA   = gmA >> pshift;
    const int remA = gmA & pmask;
    const int oyA  = remA >> wshift;
    const int oxA  = remA & wmask;
    const float* Xrow = X + nA * Cin * HW;

    // Per-thread B slot: fixed col, 16 contiguous k.
    const int bcol = tid & 63;
    const int bkl0 = (tid >> 6) << 4;          // 0 or 16
    const int gco  = nbase + bcol;

    float aReg[32];
    float bReg[16];

    auto gatherA = [&](int k0) {
        #pragma unroll
        for (int j = 0; j < 32; ++j) {
            int kk = k0 + j;                   // wave-uniform -> SALU decompose
            float v = 0.0f;
            if (kk < K) {
                int ci, iy, ix;
                if (mode == 0) {
                    ci = (int)(((u32)kk * (u32)kmulA) >> 16);
                    int kr = kk - ci * khkw;
                    int ky = (int)(((u32)kr * (u32)kmulB) >> 16);
                    int kx = kr - ky * KW;
                    iy = oyA * stride - pad + ky;
                    ix = oxA * stride - pad + kx;
                } else {
                    ci = kk >> 2;
                    int t = kk & 3;
                    iy = oyA + (1 - pa) - (t >> 1);
                    ix = oxA + (1 - pb) - (t & 1);
                }
                if (iy >= 0 && iy < Hin && ix >= 0 && ix < Win) {
                    v = Xrow[ci * HW + iy * Win + ix];
                    if (in_relu) v = fmaxf(v, 0.0f);
                }
            }
            aReg[j] = v;
        }
    };
    auto gatherB = [&](int k0) {
        #pragma unroll
        for (int j = 0; j < 16; ++j) {
            int kk = k0 + bkl0 + j;
            float v = 0.0f;
            if (kk < K && gco < Cout) {
                if (mode == 0) {
                    v = W[gco * K + kk];       // OIHW flat
                } else {
                    int ci = kk >> 2;
                    int t  = kk & 3;
                    int ky = pa + 2 * (t >> 1);
                    int kx = pb + 2 * (t & 1);
                    v = W[((ci * Cout + gco) * 4 + ky) * 4 + kx];
                }
            }
            bReg[j] = v;
        }
    };
    auto stage = [&]() {
        v8bf ap[4];
        #pragma unroll
        for (int j = 0; j < 32; ++j) ap[j >> 3][j & 7] = (bf16)aReg[j];
        v8bf* pA = (v8bf*)&As[tid][0];
        #pragma unroll
        for (int i = 0; i < 4; ++i) pA[i] = ap[i];
        v8bf bp[2];
        #pragma unroll
        for (int j = 0; j < 16; ++j) bp[j >> 3][j & 7] = (bf16)bReg[j];
        v8bf* pB = (v8bf*)&Bst[bcol][bkl0];
        pB[0] = bp[0];
        pB[1] = bp[1];
    };

    v8f acc[4][2] = {};

    gatherA(0);
    gatherB(0);
    stage();

    for (int s = 0; s < steps; ++s) {
        __syncthreads();                       // tile s visible
        const int k0n = (s + 1) << 5;
        if (s + 1 < steps) { gatherA(k0n); gatherB(k0n); }  // overlap w/ WMMA

        // Fragments: two contiguous 16B LDS loads each (CDNA5 16-bit layouts).
        const int kb   = hhalf << 3;           // A: lanes 16-31 start at K=8
        const int koff = hhalf << 3;           // B: lanes 16-31 start at K=8 (same as A)
        v16bf afrag[4];
        #pragma unroll
        for (int ms = 0; ms < 4; ++ms) {
            const v8bf* pa = (const v8bf*)&As[wm + ms * 16 + lr][kb];
            v8bf lo = pa[0];                   // k = kb .. kb+7
            v8bf hi = pa[2];                   // k = kb+16 .. kb+23
            afrag[ms] = __builtin_shufflevector(lo, hi,
                0,1,2,3,4,5,6,7,8,9,10,11,12,13,14,15);
        }
        v16bf bfrag[2];
        #pragma unroll
        for (int ns = 0; ns < 2; ++ns) {
            const v8bf* pb = (const v8bf*)&Bst[wn + ns * 16 + lr][koff];
            v8bf lo = pb[0];
            v8bf hi = pb[2];
            bfrag[ns] = __builtin_shufflevector(lo, hi,
                0,1,2,3,4,5,6,7,8,9,10,11,12,13,14,15);
        }
        #pragma unroll
        for (int ms = 0; ms < 4; ++ms)
            #pragma unroll
            for (int ns = 0; ns < 2; ++ns)
                acc[ms][ns] = __builtin_amdgcn_wmma_f32_16x16x32_bf16(
                    false, afrag[ms], false, bfrag[ns],
                    (short)0, acc[ms][ns], false, false);

        __syncthreads();                       // done reading tile s
        if (s + 1 < steps) stage();            // write tile s+1
    }

    // Epilogue: D layout = col (l&15), rows r + 8*(l>=16).
    #pragma unroll
    for (int ns = 0; ns < 2; ++ns) {
        int co = nbase + wn + ns * 16 + lr;
        if (co >= Cout) continue;
        float bv = Bias ? Bias[co] : 0.0f;
        #pragma unroll
        for (int ms = 0; ms < 4; ++ms) {
            int m0 = mbase + wm + ms * 16 + (hhalf << 3);
            #pragma unroll
            for (int r = 0; r < 8; ++r) {
                int gm  = m0 + r;
                int n   = gm >> pshift;
                int rem = gm & pmask;
                int oy  = rem >> wshift;
                int ox  = rem & wmask;
                int ho, wo;
                if (mode == 0) { ho = oy; wo = ox; }
                else { ho = 2 * oy + (1 - pa); wo = 2 * ox + (1 - pb); }
                int oidx = ((n * Cout + co) * Hout + ho) * Wout + wo;
                float v = acc[ms][ns][r] + bv;
                if (Res) v += Res[oidx];
                if (out_relu) v = fmaxf(v, 0.0f);
                Out[oidx] = v;
            }
        }
    }
}

// ---------------------------------------------------------------------------
// VQ argmin over 512 codes via f32 WMMA:  argmin_c ( ||c||^2 - 2 z.c ).
// Block = 128 threads / 4 waves = 128 vectors; codebook chunked in LDS.
// Keys are order-preserving (float->u32) << 32 | col  -> first-min tiebreak.
// ---------------------------------------------------------------------------
__global__ __launch_bounds__(128) void vqvae_vq_argmin(
    const float* __restrict__ z, const float* __restrict__ cb,
    int* __restrict__ bestidx, float* __restrict__ bestdist)
{
    __shared__ float Cs[128][68];              // chunk: 128 codes x 64 dims
    __shared__ float cns[128];                 // chunk code norms
    __shared__ float rns[128];                 // row norms
    __shared__ u64   keys[128][17];            // per-row per-lane best keys

    const int tid   = threadIdx.x;
    const int lane  = tid & 31;
    const int wave  = tid >> 5;
    const int lr    = lane & 15;
    const int hhalf = lane >> 4;
    const int rowbase = blockIdx.x << 7;

    // row norm (coalesced: consecutive tid -> consecutive hw)
    {
        int gm = rowbase + tid;
        int n = gm >> 12, hw = gm & 4095;
        const float* zp = z + n * 262144 + hw; // NCHW stride 4096 per channel
        float s = 0.0f;
        #pragma unroll
        for (int d = 0; d < 64; ++d) { float v = zp[d * 4096]; s = fmaf(v, v, s); }
        rns[tid] = s;
    }

    // A fragments for this wave's 32 rows, all 16 K-steps (K=64, 4 per step).
    v2f afrag[2][16];
    #pragma unroll
    for (int ms = 0; ms < 2; ++ms) {
        int gm = rowbase + wave * 32 + ms * 16 + lr;
        int n = gm >> 12, hw = gm & 4095;
        const float* zp = z + n * 262144 + hw;
        #pragma unroll
        for (int ks = 0; ks < 16; ++ks) {
            int k = ks * 4 + 2 * hhalf;        // A 32-bit layout: k = e + 2*(l>=16)
            afrag[ms][ks][0] = zp[k * 4096];
            afrag[ms][ks][1] = zp[(k + 1) * 4096];
        }
    }

    u64 best[2][8];
    #pragma unroll
    for (int ms = 0; ms < 2; ++ms)
        #pragma unroll
        for (int r = 0; r < 8; ++r) best[ms][r] = ~0ull;

    for (int c0 = 0; c0 < 512; c0 += 128) {
        __syncthreads();
        // stage 128 codes + norms
        {
            const float* cp = cb + (c0 + tid) * 64;
            float s = 0.0f;
            #pragma unroll
            for (int d = 0; d < 64; ++d) {
                float v = cp[d];
                Cs[tid][d] = v;
                s = fmaf(v, v, s);
            }
            cns[tid] = s;
        }
        __syncthreads();

        for (int nb = 0; nb < 4; ++nb) {       // 32 cols per pass
            v8f acc[2][2] = {};
            #pragma unroll
            for (int ks = 0; ks < 16; ++ks) {
                v2f bfrag[2];
                #pragma unroll
                for (int ns = 0; ns < 2; ++ns) {
                    int col = nb * 32 + ns * 16 + lr;
                    int k = ks * 4 + 2 * hhalf;
                    bfrag[ns][0] = Cs[col][k];
                    bfrag[ns][1] = Cs[col][k + 1];
                }
                #pragma unroll
                for (int ms = 0; ms < 2; ++ms)
                    #pragma unroll
                    for (int ns = 0; ns < 2; ++ns)
                        acc[ms][ns] = __builtin_amdgcn_wmma_f32_16x16x4_f32(
                            false, afrag[ms][ks], false, bfrag[ns],
                            (short)0, acc[ms][ns], false, false);
            }
            #pragma unroll
            for (int ns = 0; ns < 2; ++ns) {
                int coll = nb * 32 + ns * 16 + lr;
                float cn = cns[coll];
                u32 gcol = (u32)(c0 + coll);
                #pragma unroll
                for (int ms = 0; ms < 2; ++ms)
                    #pragma unroll
                    for (int r = 0; r < 8; ++r) {
                        float score = cn - 2.0f * acc[ms][ns][r];
                        u32 b = __float_as_uint(score);
                        u32 k32 = (b & 0x80000000u) ? ~b : (b | 0x80000000u);
                        u64 key = ((u64)k32 << 32) | gcol;
                        if (key < best[ms][r]) best[ms][r] = key;
                    }
            }
        }
    }

    __syncthreads();
    #pragma unroll
    for (int ms = 0; ms < 2; ++ms)
        #pragma unroll
        for (int r = 0; r < 8; ++r) {
            int rl = wave * 32 + ms * 16 + (hhalf << 3) + r;
            keys[rl][lr] = best[ms][r];
        }
    __syncthreads();

    // one thread per row: 16-way min
    u64 kmin = keys[tid][0];
    #pragma unroll
    for (int i = 1; i < 16; ++i) { u64 k = keys[tid][i]; if (k < kmin) kmin = k; }
    u32 hi = (u32)(kmin >> 32);
    u32 fb = (hi & 0x80000000u) ? (hi & 0x7FFFFFFFu) : ~hi;
    float score = __uint_as_float(fb);
    int gm = rowbase + tid;
    bestidx[gm]  = (int)(kmin & 0xFFFFFFFFu);
    bestdist[gm] = fmaxf(rns[tid] + score, 0.0f);
}

// gather quantized rows, histogram, deterministic per-block MSE partials
__global__ __launch_bounds__(256) void vqvae_vq_apply(
    const float* __restrict__ cb, const int* __restrict__ bestidx,
    const float* __restrict__ bestdist, float* __restrict__ q,
    float* __restrict__ partial, int* __restrict__ counts)
{
    __shared__ float red[256];
    const int v = blockIdx.x * 256 + threadIdx.x;
    const int n = v >> 12, hw = v & 4095;
    const int base = n * 262144 + hw;
    const int bi = bestidx[v];
    #pragma unroll
    for (int d = 0; d < 64; ++d) q[base + d * 4096] = cb[bi * 64 + d];
    atomicAdd(&counts[bi], 1);
    red[threadIdx.x] = bestdist[v];
    __syncthreads();
    for (int o = 128; o > 0; o >>= 1) {
        if (threadIdx.x < o) red[threadIdx.x] += red[threadIdx.x + o];
        __syncthreads();
    }
    if (threadIdx.x == 0) partial[blockIdx.x] = red[0];
}

__global__ void vqvae_zero(int* __restrict__ counts)
{
    counts[threadIdx.x] = 0;
}

__global__ void vqvae_finalize(const float* __restrict__ partial,
                               const int* __restrict__ counts,
                               float* __restrict__ out_loss,
                               float* __restrict__ out_perp)
{
    __shared__ float s1[512], s2[512];
    const int t = threadIdx.x;                 // 512 threads
    float p = (float)counts[t] * (1.0f / 131072.0f);
    s1[t] = partial[t];
    s2[t] = p * logf(p + 1e-10f);
    __syncthreads();
    for (int o = 256; o > 0; o >>= 1) {
        if (t < o) { s1[t] += s1[t + o]; s2[t] += s2[t + o]; }
        __syncthreads();
    }
    if (t == 0) {
        // forward: q_latent_loss == e_latent_loss numerically -> 1.25 * mse
        *out_loss = 1.25f * s1[0] / 8388608.0f;
        *out_perp = expf(-s2[0]);
    }
}

// ---------------------------------------------------------------------------
// Host side
// ---------------------------------------------------------------------------
static void launch_conv(const float* X, const float* W, const float* Bias,
                        const float* Res, float* Out,
                        int N, int Cin, int Hin, int Win,
                        int Cout, int Hout, int Wout,
                        int KH, int KW, int stride, int pad,
                        int mode, int pa, int pb, int in_relu, int out_relu,
                        hipStream_t s)
{
    int khkw = KH * KW;
    int K     = mode ? (Cin * 4) : (Cin * khkw);
    int steps = (K + 31) >> 5;
    int kmulA = (65536 + khkw - 1) / khkw;     // exact multiply-shift divide
    int kmulB = (65536 + KW - 1) / KW;
    int hq = mode ? (Hout >> 1) : Hout;
    int wq = mode ? (Wout >> 1) : Wout;
    int plane = hq * wq;                       // power of two for all layers
    int pshift = __builtin_ctz(plane);
    int wshift = __builtin_ctz(wq);
    long M = (long)N * plane;                  // multiple of 128 for all layers
    dim3 grid((unsigned)(M / 128), (unsigned)((Cout + 63) / 64));
    vqvae_conv_wmma<<<grid, 128, 0, s>>>(
        X, W, Bias, Res, Out,
        Cin, Hin, Win, Cout, Hout, Wout, KW, stride, pad, K, steps,
        khkw, kmulA, kmulB,
        pshift, plane - 1, wshift, wq - 1, mode, pa, pb, in_relu, out_relu);
}

extern "C" void kernel_launch(void* const* d_in, const int* in_sizes, int n_in,
                              void* d_out, int out_size, void* d_ws, size_t ws_size,
                              hipStream_t stream)
{
    (void)in_sizes; (void)n_in; (void)ws_size;
    const float* x    = (const float*)d_in[0];
    const float* e1w  = (const float*)d_in[1];
    const float* e1b  = (const float*)d_in[2];
    const float* e2w  = (const float*)d_in[3];
    const float* e2b  = (const float*)d_in[4];
    const float* er11 = (const float*)d_in[5];
    const float* er12 = (const float*)d_in[6];
    const float* er21 = (const float*)d_in[7];
    const float* er22 = (const float*)d_in[8];
    const float* pvw  = (const float*)d_in[9];
    const float* pvb  = (const float*)d_in[10];
    const float* cb   = (const float*)d_in[11];
    const float* d1w  = (const float*)d_in[12];
    const float* d1b  = (const float*)d_in[13];
    const float* dr11 = (const float*)d_in[14];
    const float* dr12 = (const float*)d_in[15];
    const float* dr21 = (const float*)d_in[16];
    const float* dr22 = (const float*)d_in[17];
    const float* t1w  = (const float*)d_in[18];
    const float* t1b  = (const float*)d_in[19];
    const float* t2w  = (const float*)d_in[20];
    const float* t2b  = (const float*)d_in[21];

    float* out   = (float*)d_out;
    float* recon = out + 1;                       // (32,3,256,256)

    // workspace layout (floats)
    float* B0 = (float*)d_ws;                     // 33554432 (32,64,128,128)
    float* B1 = B0 + 33554432;                    // 16777216 (32,128,64,64)
    float* B2 = B1 + 16777216;
    float* B3 = B2 + 16777216;
    float* B4 = B3 + 16777216;                    // zq (32,64,64,64) 8388608
    float* B5 = B4 + 8388608;                     // quantized
    float* partial  = B5 + 8388608;               // 512 floats
    int*   counts   = (int*)(partial + 512);      // 512 ints
    int*   bestidx  = counts + 512;               // 131072 ints
    float* bestdist = (float*)(bestidx + 131072); // 131072 floats

    // ---------------- encoder ----------------
    launch_conv(x, e1w, e1b, nullptr, B0, 32, 3, 256, 256, 64, 128, 128,
                4, 4, 2, 1, 0, 0, 0, 0, 1, stream);
    launch_conv(B0, e2w, e2b, nullptr, B1, 32, 64, 128, 128, 128, 64, 64,
                4, 4, 2, 1, 0, 0, 0, 0, 0, stream);
    launch_conv(B1, er11, nullptr, nullptr, B2, 32, 128, 64, 64, 128, 64, 64,
                3, 3, 1, 1, 0, 0, 0, 1, 0, stream);
    launch_conv(B2, er12, nullptr, B1, B3, 32, 128, 64, 64, 128, 64, 64,
                3, 3, 1, 1, 0, 0, 0, 1, 1, stream);
    launch_conv(B3, er21, nullptr, nullptr, B2, 32, 128, 64, 64, 128, 64, 64,
                3, 3, 1, 1, 0, 0, 0, 1, 0, stream);
    launch_conv(B2, er22, nullptr, B3, B1, 32, 128, 64, 64, 128, 64, 64,
                3, 3, 1, 1, 0, 0, 0, 1, 1, stream);
    launch_conv(B1, pvw, pvb, nullptr, B4, 32, 128, 64, 64, 64, 64, 64,
                1, 1, 1, 0, 0, 0, 0, 0, 0, stream);

    // ---------------- vector quantization ----------------
    vqvae_zero<<<1, 512, 0, stream>>>(counts);
    vqvae_vq_argmin<<<1024, 128, 0, stream>>>(B4, cb, bestidx, bestdist);
    vqvae_vq_apply<<<512, 256, 0, stream>>>(cb, bestidx, bestdist, B5,
                                            partial, counts);

    // ---------------- decoder ----------------
    launch_conv(B5, d1w, d1b, nullptr, B1, 32, 64, 64, 64, 128, 64, 64,
                3, 3, 1, 1, 0, 0, 0, 0, 0, stream);
    launch_conv(B1, dr11, nullptr, nullptr, B2, 32, 128, 64, 64, 128, 64, 64,
                3, 3, 1, 1, 0, 0, 0, 1, 0, stream);
    launch_conv(B2, dr12, nullptr, B1, B3, 32, 128, 64, 64, 128, 64, 64,
                3, 3, 1, 1, 0, 0, 0, 1, 1, stream);
    launch_conv(B3, dr21, nullptr, nullptr, B2, 32, 128, 64, 64, 128, 64, 64,
                3, 3, 1, 1, 0, 0, 0, 1, 0, stream);
    launch_conv(B2, dr22, nullptr, B3, B1, 32, 128, 64, 64, 128, 64, 64,
                3, 3, 1, 1, 0, 0, 0, 1, 1, stream);
    // tconv1 (128 -> 64, 64x64 -> 128x128) + relu; 4 output parity classes
    for (int pa = 0; pa < 2; ++pa)
        for (int pb = 0; pb < 2; ++pb)
            launch_conv(B1, t1w, t1b, nullptr, B0, 32, 128, 64, 64, 64, 128, 128,
                        4, 4, 2, 1, 1, pa, pb, 0, 1, stream);
    // tconv2 (64 -> 3, 128x128 -> 256x256) -> x_recon
    for (int pa = 0; pa < 2; ++pa)
        for (int pb = 0; pb < 2; ++pb)
            launch_conv(B0, t2w, t2b, nullptr, recon, 32, 64, 128, 128, 3, 256, 256,
                        4, 4, 2, 1, 1, pa, pb, 0, 0, stream);

    // ---------------- scalars ----------------
    vqvae_finalize<<<1, 512, 0, stream>>>(partial, counts,
                                          out, out + (out_size - 1));
}
